// Model_63213328662715
// MI455X (gfx1250) — matrix operand order repaired and checked by reading, not compiled, not run on hardware
//
#include <hip/hip_runtime.h>
#include <hip/hip_bf16.h>
#include <math.h>

// ---------------------------------------------------------------------------
// Model constants (match reference)
// ---------------------------------------------------------------------------
static constexpr int N_    = 50000;
static constexpr int DIN_  = 128;
static constexpr int H_    = 128;
static constexpr int HEADS_= 4;
static constexpr int DH_   = 32;
static constexpr int KHOP_ = 5;
static constexpr int DOUT_ = 2;
static constexpr long E_   = 600000;
static constexpr long EL_  = 200000;

typedef __bf16 bf16_t;
typedef __attribute__((ext_vector_type(16))) __bf16 v16bf;
typedef __attribute__((ext_vector_type(8)))  __bf16 v8bf;
typedef __attribute__((ext_vector_type(8)))  float  v8f;

#define DEV_INLINE __device__ __forceinline__

// Ordered-uint encoding of float for atomicMax-based float max
DEV_INLINE unsigned fenc(float f) {
    unsigned u = __float_as_uint(f);
    return (u & 0x80000000u) ? ~u : (u | 0x80000000u);
}
DEV_INLINE float fdec(unsigned k) {
    return (k & 0x80000000u) ? __uint_as_float(k & 0x7FFFFFFFu)
                             : __uint_as_float(~k);
}

// ---------------------------------------------------------------------------
// Utility kernels
// ---------------------------------------------------------------------------
__global__ void fill_u32_kernel(unsigned* __restrict__ p, unsigned v, long n) {
    long i = (long)blockIdx.x * blockDim.x + threadIdx.x;
    if (i < n) p[i] = v;
}

__global__ void cvt_bf16_kernel(const float* __restrict__ s, bf16_t* __restrict__ d, long n) {
    long i = (long)blockIdx.x * blockDim.x + threadIdx.x;
    if (i < n) d[i] = (bf16_t)s[i];
}

// Pack f32 weight B[K][Nc] into bf16 WMMA-fragment order:
//   Bp[((c*ntn + tn)*32 + lane)*16 + e]  with  k = c*32 + (lane>>4)*16 + e,
//   n = tn*16 + (lane&15).  GEMM then loads B frags as one contiguous v16bf.
__global__ void pack_b_kernel(const float* __restrict__ B, bf16_t* __restrict__ Bp,
                              int K, int Nc) {
    long total = (long)K * Nc;
    long i = (long)blockIdx.x * blockDim.x + threadIdx.x;
    if (i >= total) return;
    int  e    = (int)(i & 15);
    long r    = i >> 4;
    int  lane = (int)(r & 31);
    long ct   = r >> 5;
    int  ntn  = Nc >> 4;
    int  tn   = (int)(ct % ntn);
    int  c    = (int)(ct / ntn);
    int  k    = c * 32 + (lane >> 4) * 16 + e;
    int  n    = tn * 16 + (lane & 15);
    Bp[i] = (bf16_t)B[(size_t)k * Nc + n];
}

// Min over non-negative floats via uint-bit comparison (edge_feature in [0,1))
__global__ void min_reduce_kernel(const float* __restrict__ t, unsigned* __restrict__ out, long n) {
    long i = (long)blockIdx.x * blockDim.x + threadIdx.x;
    long stride = (long)gridDim.x * blockDim.x;
    unsigned local = 0x7F7FFFFFu; // FLT_MAX bits
    for (; i < n; i += stride) {
        unsigned u = __float_as_uint(t[i]);
        local = (u < local) ? u : local;
    }
    atomicMin(out, local);
}

// te[e,j] = cos((t[e]-tmin+1) * basis[j]), stored bf16
__global__ void time_encode_kernel(const float* __restrict__ tf, const float* __restrict__ basis,
                                   const unsigned* __restrict__ tminb, bf16_t* __restrict__ te,
                                   long total) {
    long i = (long)blockIdx.x * blockDim.x + threadIdx.x;
    if (i >= total) return;
    long e = i >> 7;
    int  j = (int)(i & 127);
    float t = tf[e] - __uint_as_float(*tminb) + 1.0f;
    te[i] = (bf16_t)cosf(t * basis[j]);
}

// ---------------------------------------------------------------------------
// bf16 WMMA GEMM, 16x128 output stripe per wave (8 independent accumulators):
//   C[M,Nc] = op(A[M,K] @ B[K,Nc] + bias) (+ resid), optional bf16 copy-out.
// Requires Nc % 128 == 0, M % 16 == 0, K % 32 == 0.
// Per K-chunk: 1 A-frag load (2 x b128) + 8 contiguous B-frag loads + 8 WMMAs
// on independent chains -> 8x less A traffic, latency-hiding WMMA pipeline.
// A-frag (16-bit 16x32): lane row = lane&15, khalf = lane>>4;
//   elem i -> k=khalf*16+i (matches packed B): two 16B loads.
// B-frag: contiguous v16bf from packed Bp (see pack_b_kernel).
// C: vgpr j -> row j + 8*khalf, col lane&15.
// ---------------------------------------------------------------------------
__global__ __launch_bounds__(128)
void gemm_bf16_wmma8(const bf16_t* __restrict__ A, const bf16_t* __restrict__ Bp,
                     const float* __restrict__ bias, const float* __restrict__ resid,
                     float* __restrict__ C, bf16_t* __restrict__ Cbf,
                     int M, int K, int Nc, int relu) {
    int wave = blockIdx.x * (blockDim.x >> 5) + (threadIdx.x >> 5);
    int lane = threadIdx.x & 31;
    int ntn     = Nc >> 4;   // 16-wide N tiles
    int ngroups = ntn >> 3;  // groups of 8 N tiles (128 cols)
    int total   = (M >> 4) * ngroups;
    if (wave >= total) return;
    int tm = wave / ngroups;
    int tg = wave % ngroups;
    int m     = lane & 15;
    int khalf = lane >> 4;
    int mrow  = (tm << 4) + m;

    v8f acc[8];
    #pragma unroll
    for (int t = 0; t < 8; ++t)
        #pragma unroll
        for (int i = 0; i < 8; ++i) acc[t][i] = 0.0f;

    const bf16_t* ap = A  + (size_t)mrow * K + khalf * 16;
    const bf16_t* bp = Bp + (((size_t)tg * 8) * 32 + lane) * 16;
    const size_t  bstride = (size_t)ntn * 512;   // per k-chunk advance in Bp

    const int nchunks = K >> 5;
    for (int c = 0; c < nchunks; ++c) {
        v8bf alo = *(const v8bf*)(ap);
        v8bf ahi = *(const v8bf*)(ap + 8);
        v16bf a = __builtin_shufflevector(alo, ahi,
                    0,1,2,3,4,5,6,7,8,9,10,11,12,13,14,15);
        #pragma unroll
        for (int t = 0; t < 8; ++t) {
            v16bf b = *(const v16bf*)(bp + (size_t)t * 512);
            acc[t] = __builtin_amdgcn_wmma_f32_16x16x32_bf16(
                         false, a, false, b, (short)0, acc[t], false, false);
        }
        ap += 32;
        bp += bstride;
    }

    #pragma unroll
    for (int t = 0; t < 8; ++t) {
        int ncol = (tg << 7) + (t << 4) + m;
        float bv = bias ? bias[ncol] : 0.0f;
        #pragma unroll
        for (int j = 0; j < 8; ++j) {
            int r = (tm << 4) + j + khalf * 8;
            float v = acc[t][j] + bv;
            if (relu) v = fmaxf(v, 0.0f);
            size_t o = (size_t)r * Nc + ncol;
            if (resid) v += resid[o];
            C[o] = v;
            if (Cbf) Cbf[o] = (bf16_t)v;
        }
    }
}

// ---------------------------------------------------------------------------
// Attention pass 1: per (edge, head) score + atomic segment max over dst
// ---------------------------------------------------------------------------
__global__ void attn_score_kernel(const float* __restrict__ Q, const float* __restrict__ Km,
                                  const float* __restrict__ ep, const int* __restrict__ src,
                                  const int* __restrict__ dst, float* __restrict__ sb,
                                  unsigned* __restrict__ mkey, long total, float scale) {
    long idx = (long)blockIdx.x * blockDim.x + threadIdx.x;
    if (idx >= total) return;
    long e  = idx >> 2;
    int  hd = (int)(idx & 3);
    int sn = src[e], dn = dst[e];
    const float4* qp  = (const float4*)(Q  + (size_t)dn * H_ + hd * DH_);
    const float4* kp  = (const float4*)(Km + (size_t)sn * H_ + hd * DH_);
    const float4* epp = (const float4*)(ep + (size_t)e  * H_ + hd * DH_);
    float acc = 0.0f;
    #pragma unroll
    for (int i = 0; i < 8; ++i) {
        float4 q4 = qp[i], k4 = kp[i], e4 = epp[i];
        acc += q4.x * (k4.x + e4.x) + q4.y * (k4.y + e4.y)
             + q4.z * (k4.z + e4.z) + q4.w * (k4.w + e4.w);
    }
    acc *= scale;
    sb[idx] = acc;
    atomicMax(&mkey[(size_t)dn * HEADS_ + hd], fenc(acc));
}

// ---------------------------------------------------------------------------
// Attention pass 2: w = exp(s - m); accumulate z and numerator
// ---------------------------------------------------------------------------
__global__ void attn_accum_kernel(const float* __restrict__ V, const float* __restrict__ ep,
                                  const float* __restrict__ sb, const unsigned* __restrict__ mkey,
                                  const int* __restrict__ src, const int* __restrict__ dst,
                                  float* __restrict__ z, float* __restrict__ num, long total) {
    long idx = (long)blockIdx.x * blockDim.x + threadIdx.x;
    if (idx >= total) return;
    long e  = idx >> 2;
    int  hd = (int)(idx & 3);
    int sn = src[e], dn = dst[e];
    float w = expf(sb[idx] - fdec(mkey[(size_t)dn * HEADS_ + hd]));
    atomicAdd(&z[(size_t)dn * HEADS_ + hd], w);
    const float4* vp  = (const float4*)(V  + (size_t)sn * H_ + hd * DH_);
    const float4* epp = (const float4*)(ep + (size_t)e  * H_ + hd * DH_);
    float* op = num + (size_t)dn * H_ + hd * DH_;
    #pragma unroll
    for (int i = 0; i < 8; ++i) {
        float4 v4 = vp[i], e4 = epp[i];
        atomicAdd(op + 4 * i + 0, w * (v4.x + e4.x));
        atomicAdd(op + 4 * i + 1, w * (v4.y + e4.y));
        atomicAdd(op + 4 * i + 2, w * (v4.z + e4.z));
        atomicAdd(op + 4 * i + 3, w * (v4.w + e4.w));
    }
}

// agg = num / (z + 1e-16) -> bf16 (input for Wo GEMM)
__global__ void attn_norm_kernel(const float* __restrict__ num, const float* __restrict__ z,
                                 bf16_t* __restrict__ aggb, long total) {
    long i = (long)blockIdx.x * blockDim.x + threadIdx.x;
    if (i >= total) return;
    long n = i >> 7;
    int  j = (int)(i & 127);
    float zz = z[(size_t)n * HEADS_ + (j >> 5)] + 1e-16f;
    aggb[i] = (bf16_t)(num[i] / zz);
}

// ---------------------------------------------------------------------------
// GRU elementwise (torch semantics), writes hnew (f32 out) + bf16 copy
// ---------------------------------------------------------------------------
__global__ void gru_kernel(const float* __restrict__ gi, const float* __restrict__ gh,
                           const float* __restrict__ prev, float* __restrict__ hnew,
                           bf16_t* __restrict__ hnewbf, long total) {
    long i = (long)blockIdx.x * blockDim.x + threadIdx.x;
    if (i >= total) return;
    long n = i >> 7;
    int  j = (int)(i & 127);
    const float* gir = gi + (size_t)n * (3 * H_);
    const float* ghr = gh + (size_t)n * (3 * H_);
    float r  = 1.0f / (1.0f + expf(-(gir[j] + ghr[j])));
    float zg = 1.0f / (1.0f + expf(-(gir[H_ + j] + ghr[H_ + j])));
    float nn = tanhf(gir[2 * H_ + j] + r * ghr[2 * H_ + j]);
    float p  = prev[i];
    float h  = (1.0f - zg) * nn + zg * p;
    hnew[i]   = h;
    hnewbf[i] = (bf16_t)h;
}

// ---------------------------------------------------------------------------
// Link decoder: one block (4 waves) per 16 label edges.
// dh = relu([h[e0] | h[e1]] @ Wd1 + bd1) via gathered WMMA over K=256
// (Wd1 fragment-packed), then pred = dh @ Wd2 + bd2 through LDS.
// ---------------------------------------------------------------------------
__global__ __launch_bounds__(128)
void decoder_kernel(const bf16_t* __restrict__ hbf, const bf16_t* __restrict__ Wd1p,
                    const float* __restrict__ bd1, const float* __restrict__ Wd2,
                    const float* __restrict__ bd2, const int* __restrict__ e0,
                    const int* __restrict__ e1, float* __restrict__ pred) {
    __shared__ float dh[16][128];
    int tile  = blockIdx.x;
    int lane  = threadIdx.x & 31;
    int wv    = threadIdx.x >> 5;
    int m     = lane & 15;
    int khalf = lane >> 4;
    int el    = tile * 16 + m;
    int n0    = e0[el];
    int n1    = e1[el];

    for (int t = 0; t < 2; ++t) {
        int tn   = wv * 2 + t;
        int ncol = tn * 16 + m;
        v8f acc;
        #pragma unroll
        for (int i = 0; i < 8; ++i) acc[i] = 0.0f;
        #pragma unroll
        for (int c = 0; c < 8; ++c) {                 // K = 256, 8 chunks
            int k0  = c * 32;
            int nid = (k0 < H_) ? n0 : n1;
            const bf16_t* ap = hbf + (size_t)nid * H_ + (k0 & (H_ - 1)) + khalf * 16;
            v8bf alo = *(const v8bf*)(ap);
            v8bf ahi = *(const v8bf*)(ap + 8);
            v16bf a = __builtin_shufflevector(alo, ahi,
                        0,1,2,3,4,5,6,7,8,9,10,11,12,13,14,15);
            v16bf b = *(const v16bf*)(Wd1p + (((size_t)c * 8 + tn) * 32 + lane) * 16);
            acc = __builtin_amdgcn_wmma_f32_16x16x32_bf16(
                      false, a, false, b, (short)0, acc, false, false);
        }
        #pragma unroll
        for (int j = 0; j < 8; ++j) {
            int r = j + khalf * 8;
            float v = acc[j] + bd1[ncol];
            dh[r][ncol] = fmaxf(v, 0.0f);
        }
    }
    __syncthreads();
    if (threadIdx.x < 32) {
        int r = threadIdx.x >> 1;
        int c = threadIdx.x & 1;
        float a = bd2[c];
        #pragma unroll 4
        for (int j = 0; j < H_; ++j) a += dh[r][j] * Wd2[j * DOUT_ + c];
        pred[(size_t)(tile * 16 + r) * DOUT_ + c] = a;
    }
}

// ---------------------------------------------------------------------------
// Host driver
// ---------------------------------------------------------------------------
extern "C" void kernel_launch(void* const* d_in, const int* in_sizes, int n_in,
                              void* d_out, int out_size, void* d_ws, size_t ws_size,
                              hipStream_t stream) {
    (void)in_sizes; (void)n_in; (void)out_size; (void)ws_size;

    const float* x      = (const float*)d_in[0];
    const float* efeat  = (const float*)d_in[1];
    const float* prev   = (const float*)d_in[2];
    const float* W_mlp  = (const float*)d_in[3];
    const float* b_mlp  = (const float*)d_in[4];
    const float* basis  = (const float*)d_in[5];
    const float* Wq     = (const float*)d_in[6];
    const float* Wk     = (const float*)d_in[7];
    const float* Wv     = (const float*)d_in[8];
    const float* We     = (const float*)d_in[9];
    const float* Wo     = (const float*)d_in[10];
    const float* W_ih   = (const float*)d_in[11];
    const float* W_hh   = (const float*)d_in[12];
    const float* b_ih   = (const float*)d_in[13];
    const float* b_hh   = (const float*)d_in[14];
    const float* Wd1    = (const float*)d_in[15];
    const float* bd1    = (const float*)d_in[16];
    const float* Wd2    = (const float*)d_in[17];
    const float* bd2    = (const float*)d_in[18];
    const int*   eidx   = (const int*)d_in[19];
    const int*   elidx  = (const int*)d_in[20];

    const int* src  = eidx;
    const int* dst  = eidx + E_;
    const int* eli0 = elidx;
    const int* eli1 = elidx + EL_;

    float* pred = (float*)d_out;                 // [EL, 2]
    float* hnew = pred + EL_ * DOUT_;            // [N, H]

    // Workspace bump allocator (256B aligned)
    char* wp = (char*)d_ws;
    auto bump = [&](size_t bytes) -> void* {
        void* r = wp;
        wp += (bytes + 255) & ~(size_t)255;
        return r;
    };

    // Fragment-packed bf16 weights
    bf16_t* Wmlp_p = (bf16_t*)bump((size_t)DIN_ * H_ * 2);
    bf16_t* Wq_p   = (bf16_t*)bump((size_t)H_ * H_ * 2);
    bf16_t* Wk_p   = (bf16_t*)bump((size_t)H_ * H_ * 2);
    bf16_t* Wv_p   = (bf16_t*)bump((size_t)H_ * H_ * 2);
    bf16_t* We_p   = (bf16_t*)bump((size_t)H_ * H_ * 2);
    bf16_t* Wo_p   = (bf16_t*)bump((size_t)H_ * H_ * 2);
    bf16_t* Wih_p  = (bf16_t*)bump((size_t)H_ * 3 * H_ * 2);
    bf16_t* Whh_p  = (bf16_t*)bump((size_t)H_ * 3 * H_ * 2);
    bf16_t* Wd1_p  = (bf16_t*)bump((size_t)2 * H_ * H_ * 2);
    // Activations / buffers
    bf16_t* x_b    = (bf16_t*)bump((size_t)N_ * DIN_ * 2);
    bf16_t* prev_b = (bf16_t*)bump((size_t)N_ * H_ * 2);
    float*  h      = (float*) bump((size_t)N_ * H_ * 4);
    bf16_t* h_b    = (bf16_t*)bump((size_t)N_ * H_ * 2);
    float*  Qb     = (float*) bump((size_t)N_ * H_ * 4);
    float*  Kb     = (float*) bump((size_t)N_ * H_ * 4);
    float*  Vb     = (float*) bump((size_t)N_ * H_ * 4);
    bf16_t* agg_b  = (bf16_t*)bump((size_t)N_ * H_ * 2);
    // mkey / zb / numb contiguous (all sizes multiples of 256B): one zero-fill
    unsigned* mkey = (unsigned*)bump((size_t)N_ * HEADS_ * 4);
    float*  zb     = (float*) bump((size_t)N_ * HEADS_ * 4);
    float*  numb   = (float*) bump((size_t)N_ * H_ * 4);
    float*  sbuf   = (float*) bump((size_t)E_ * HEADS_ * 4);
    bf16_t* te_b   = (bf16_t*)bump((size_t)E_ * H_ * 2);
    float*  ep     = (float*) bump((size_t)E_ * H_ * 4);
    float*  gi     = (float*) bump((size_t)N_ * 3 * H_ * 4);
    float*  gh     = (float*) bump((size_t)N_ * 3 * H_ * 4);
    bf16_t* hnew_b = (bf16_t*)bump((size_t)N_ * H_ * 2);
    unsigned* tminb= (unsigned*)bump(256);

    auto nblk = [](long n, int t) { return (int)((n + t - 1) / t); };
    auto cvt = [&](const float* s, bf16_t* d, long n) {
        cvt_bf16_kernel<<<nblk(n, 256), 256, 0, stream>>>(s, d, n);
    };
    auto pack = [&](const float* B, bf16_t* Bp, int K, int Nc) {
        pack_b_kernel<<<nblk((long)K * Nc, 256), 256, 0, stream>>>(B, Bp, K, Nc);
    };
    auto gemm = [&](const bf16_t* A, const bf16_t* Bp, const float* bias,
                    const float* resid, float* C, bf16_t* Cbf,
                    int M, int K, int Nc, int relu) {
        int waves = (M / 16) * (Nc / 128);   // one 16x128 stripe per wave
        gemm_bf16_wmma8<<<(waves + 3) / 4, 128, 0, stream>>>(
            A, Bp, bias, resid, C, Cbf, M, K, Nc, relu);
    };

    // --- Stage 0: pack weights into WMMA fragment order; convert inputs ---
    pack(W_mlp, Wmlp_p, DIN_, H_);
    pack(Wq,    Wq_p,   H_, H_);
    pack(Wk,    Wk_p,   H_, H_);
    pack(Wv,    Wv_p,   H_, H_);
    pack(We,    We_p,   H_, H_);
    pack(Wo,    Wo_p,   H_, H_);
    pack(W_ih,  Wih_p,  H_, 3 * H_);
    pack(W_hh,  Whh_p,  H_, 3 * H_);
    pack(Wd1,   Wd1_p,  2 * H_, H_);
    cvt(x,    x_b,    (long)N_ * DIN_);
    cvt(prev, prev_b, (long)N_ * H_);

    // --- Stage 1: t.min() (lexsort skipped: permutation-invariant downstream) ---
    fill_u32_kernel<<<1, 32, 0, stream>>>(tminb, 0x7F7FFFFFu, 1);
    min_reduce_kernel<<<512, 256, 0, stream>>>(efeat, tminb, E_);

    // --- Stage 2: time encoding + ep = te @ We ---
    time_encode_kernel<<<nblk(E_ * H_, 256), 256, 0, stream>>>(efeat, basis, tminb, te_b, E_ * H_);
    gemm(te_b, We_p, nullptr, nullptr, ep, nullptr, (int)E_, H_, H_, 0);

    // --- Stage 3: h = relu(x @ W_mlp + b_mlp), also emit bf16 copy ---
    gemm(x_b, Wmlp_p, b_mlp, nullptr, h, h_b, N_, DIN_, H_, 1);

    // --- Stage 4: KHOP attention hops ---
    const float scale = 0.17677669529663687f; // 1/sqrt(32)
    const long zero_words = (long)N_ * (HEADS_ + HEADS_ + H_); // mkey+zb+numb
    for (int hop = 0; hop < KHOP_; ++hop) {
        gemm(h_b, Wq_p, nullptr, nullptr, Qb, nullptr, N_, H_, H_, 0);
        gemm(h_b, Wk_p, nullptr, nullptr, Kb, nullptr, N_, H_, H_, 0);
        gemm(h_b, Wv_p, nullptr, nullptr, Vb, nullptr, N_, H_, H_, 0);

        fill_u32_kernel<<<nblk(zero_words, 256), 256, 0, stream>>>(mkey, 0u, zero_words);

        attn_score_kernel<<<nblk(E_ * HEADS_, 256), 256, 0, stream>>>(
            Qb, Kb, ep, src, dst, sbuf, mkey, E_ * HEADS_, scale);
        attn_accum_kernel<<<nblk(E_ * HEADS_, 256), 256, 0, stream>>>(
            Vb, ep, sbuf, mkey, src, dst, zb, numb, E_ * HEADS_);
        attn_norm_kernel<<<nblk((long)N_ * H_, 256), 256, 0, stream>>>(
            numb, zb, agg_b, (long)N_ * H_);

        // h = relu(agg @ Wo) + h  (resid after relu); emits next hop's bf16 h
        gemm(agg_b, Wo_p, nullptr, h, h, h_b, N_, H_, H_, 1);
    }

    // --- Stage 5: GRU ---
    gemm(h_b,    Wih_p, b_ih, nullptr, gi, nullptr, N_, H_, 3 * H_, 0);
    gemm(prev_b, Whh_p, b_hh, nullptr, gh, nullptr, N_, H_, 3 * H_, 0);
    gru_kernel<<<nblk((long)N_ * H_, 256), 256, 0, stream>>>(gi, gh, prev, hnew, hnew_b, (long)N_ * H_);

    // --- Stage 6: link decoder ---
    decoder_kernel<<<(int)(EL_ / 16), 128, 0, stream>>>(
        hnew_b, Wd1_p, bd1, Wd2, bd2, eli0, eli1, pred);
}
